// DNATransportHamiltonianGNN_48000554500510
// MI455X (gfx1250) — matrix operand order repaired and checked by reading, not compiled, not run on hardware
//
#include <hip/hip_runtime.h>

#define HDIM   128
#define NHEAD  4
#define NE_PTS 100
#define HS     128
#define NLAYER 4

typedef __attribute__((ext_vector_type(16))) _Float16 v16h;
typedef __attribute__((ext_vector_type(8)))  float    v8f;
typedef __attribute__((ext_vector_type(4)))  unsigned int v4u;
typedef __attribute__((ext_vector_type(8)))  int      v8i;
typedef __attribute__((ext_vector_type(4)))  int      v4i;

#if defined(__has_builtin)
#if __has_builtin(__builtin_amdgcn_tensor_load_to_lds) && __has_builtin(__builtin_amdgcn_s_wait_tensorcnt)
#define HAVE_TDM 1
#endif
#endif
#ifndef HAVE_TDM
#define HAVE_TDM 0
#endif

// ---------------------------------------------------------------------------
// WMMA fragment index helpers (CDNA5 wave32 layouts, ISA 7.12.2)
// A (16x32 f16): lane L holds row M=L&15; half=L>>4 selects K-subblock.
//   element j (0..15): k = (j<8?0:16) + half*8 + ((j>>1)&3)*2 + (j&1)
// B (32x16 f16): lane L holds col N=L&15; element j: k = (j<8?0:16) + half*8 + (j&7)
// C/D (16x16 f32): lane L col N=L&15; vgpr r row M = r + 8*half
// ---------------------------------------------------------------------------
__device__ __forceinline__ int klA(int j, int half) {
  return ((j < 8) ? 0 : 16) + half * 8 + (((j >> 1) & 3) << 1) + (j & 1);
}
__device__ __forceinline__ int klB(int j, int half) { return ((j < 8) ? 0 : 16) + half * 8 + (j & 7); }

__device__ __forceinline__ void atomicMaxFloat(float* addr, float val) {
  int* ia = (int*)addr;
  int old = *ia;
  while (__int_as_float(old) < val) {
    int assumed = old;
    old = atomicCAS(ia, assumed, __float_as_int(val));
    if (old == assumed) break;
  }
}

// ---------------------------------------------------------------------------
// C[M x 128] = A[M x 128] @ W[128 x 128], f16 WMMA.
// One 256-thread block (8 waves) per 16-row stripe: A tile staged to LDS in
// f16 once (8x reuse), each wave owns one 16-col tile, K loop fully unrolled.
// ---------------------------------------------------------------------------
__global__ __launch_bounds__(256) void wmma_matmul128(
    const float* __restrict__ A, const float* __restrict__ W,
    float* __restrict__ C, int M) {
  __shared__ _Float16 Ash[16 * HDIM];
  int tid  = threadIdx.x;
  int wave = tid >> 5;
  int lane = tid & 31;
  int half = lane >> 4;
  int l15  = lane & 15;
  int m0   = blockIdx.x << 4;
  int n0   = wave << 4;

  __builtin_prefetch(&W[(size_t)0], 0, 0);      // global_prefetch_b8
  // stage A tile (16x128) as f16, coalesced
  for (int idx = tid; idx < 16 * HDIM; idx += 256) {
    int r = idx >> 7, c = idx & 127;
    int row = m0 + r; if (row >= M) row = M - 1;  // clamp keeps EXEC full
    Ash[idx] = (_Float16)A[row * HDIM + c];
  }
  __syncthreads();

  v8f acc = {};
#pragma unroll
  for (int k0 = 0; k0 < HDIM; k0 += 32) {
    v16h a, b;
#pragma unroll
    for (int j = 0; j < 16; ++j) {
      a[j] = Ash[l15 * HDIM + k0 + klA(j, half)];
      b[j] = (_Float16)W[(k0 + klB(j, half)) * HDIM + n0 + l15];
    }
    acc = __builtin_amdgcn_wmma_f32_16x16x32_f16(false, a, false, b,
                                                 (short)0, acc, false, false);
  }
#pragma unroll
  for (int r = 0; r < 8; ++r) {
    int orow = m0 + r + half * 8;
    if (orow < M) C[orow * HDIM + n0 + l15] = acc[r];
  }
}

// ---------------------------------------------------------------------------
// small encoders / MLPs / attention plumbing
// ---------------------------------------------------------------------------
__global__ void k_encode(const float* __restrict__ x, const float* __restrict__ w,
                         const float* __restrict__ b, float* __restrict__ out,
                         int n, int kdim) {
  int idx = blockIdx.x * blockDim.x + threadIdx.x;
  if (idx >= n * HDIM) return;
  int i = idx >> 7, c = idx & 127;
  float acc = b[c];
  for (int k = 0; k < kdim; ++k) acc += x[i * kdim + k] * w[k * HDIM + c];
  out[idx] = acc;
}

__global__ void k_el_mean(const float* __restrict__ el, float* __restrict__ m, int E) {
  int c = threadIdx.x;
  float s = 0.f;
  for (int f = 0; f < E; ++f) s += el[f * HDIM + c];
  m[c] = s / (float)E;
}

__global__ void k_zero3(float* amax, float* den, float* hc, int N) {
  int idx = blockIdx.x * blockDim.x + threadIdx.x;
  if (idx < N * HDIM) hc[idx] = 0.f;
  if (idx < N * NHEAD) { amax[idx] = -1e30f; den[idx] = 0.f; }
}

__global__ void k_alpha(const float* __restrict__ xl, const float* __restrict__ el,
                        const float* __restrict__ elm,
                        const int* __restrict__ src, const int* __restrict__ dst,
                        const float* __restrict__ asrc, const float* __restrict__ adst,
                        const float* __restrict__ aedge,
                        float* __restrict__ ex, float* __restrict__ amax,
                        int E, int N) {
  int idx = blockIdx.x * blockDim.x + threadIdx.x;
  int F = E + N;
  if (idx >= F * NHEAD) return;
  int f = idx >> 2, hd = idx & 3;
  int s, d; const float* ev;
  if (f < E) { s = src[f]; d = dst[f]; ev = el + f * HDIM + hd * 32; }
  else       { s = d = f - E;          ev = elm + hd * 32; }
  const float* xs = xl + s * HDIM + hd * 32;
  const float* xd = xl + d * HDIM + hd * 32;
  const float* a0 = asrc + hd * 32;
  const float* a1 = adst + hd * 32;
  const float* a2 = aedge + hd * 32;
  float a = 0.f;
  for (int c = 0; c < 32; ++c) a += xs[c] * a0[c] + xd[c] * a1[c] + ev[c] * a2[c];
  a = (a > 0.f) ? a : 0.2f * a;          // leaky_relu(0.2)
  ex[idx] = a;
  atomicMaxFloat(&amax[d * NHEAD + hd], a);
}

__global__ void k_expsum(const int* __restrict__ dst, float* __restrict__ ex,
                         const float* __restrict__ amax, float* __restrict__ den,
                         int E, int N) {
  int idx = blockIdx.x * blockDim.x + threadIdx.x;
  int F = E + N;
  if (idx >= F * NHEAD) return;
  int f = idx >> 2, hd = idx & 3;
  int d = (f < E) ? dst[f] : (f - E);
  float v = expf(ex[idx] - amax[d * NHEAD + hd]);
  ex[idx] = v;
  atomicAdd(&den[d * NHEAD + hd], v);
}

__global__ void k_scatter(const float* __restrict__ xl,
                          const int* __restrict__ src, const int* __restrict__ dst,
                          const float* __restrict__ ex, const float* __restrict__ den,
                          float* __restrict__ hc, int E, int N) {
  int idx = blockIdx.x * blockDim.x + threadIdx.x;
  int F = E + N;
  if (idx >= F * HDIM) return;
  int f = idx >> 7, cc = idx & 127, hd = cc >> 5;
  int s, d;
  if (f < E) { s = src[f]; d = dst[f]; } else { s = d = f - E; }
  float w = ex[f * NHEAD + hd] / den[d * NHEAD + hd];
  atomicAdd(&hc[d * HDIM + cc], xl[s * HDIM + cc] * w);
}

__global__ __launch_bounds__(128) void k_norm(float* __restrict__ h,
                       const float* __restrict__ hc, const float* __restrict__ bias,
                       const float* __restrict__ lns, const float* __restrict__ lnb) {
  __shared__ float red[128];
  int n = blockIdx.x, t = threadIdx.x;
  float hcv = hc[n * HDIM + t] + bias[t];
  float v = (hcv > 0.f ? hcv : 0.f) + h[n * HDIM + t];
  red[t] = v; __syncthreads();
  for (int s = 64; s > 0; s >>= 1) { if (t < s) red[t] += red[t + s]; __syncthreads(); }
  float mu = red[0] / 128.f; __syncthreads();
  float dv = v - mu;
  red[t] = dv * dv; __syncthreads();
  for (int s = 64; s > 0; s >>= 1) { if (t < s) red[t] += red[t + s]; __syncthreads(); }
  float var = red[0] / 128.f;
  h[n * HDIM + t] = dv * rsqrtf(var + 1e-5f) * lns[t] + lnb[t];
}

__global__ __launch_bounds__(64) void k_mlp(const float* __restrict__ in,
                      const float* __restrict__ w1, const float* __restrict__ b1,
                      const float* __restrict__ w2, const float* __restrict__ b2,
                      float* __restrict__ out) {
  __shared__ float red[64];
  int i = blockIdx.x, t = threadIdx.x;
  float a = b1[t];
  const float* iv = in + i * HDIM;
  for (int k = 0; k < HDIM; ++k) a += iv[k] * w1[k * 64 + t];
  a = (a > 0.f) ? a : 0.f;
  red[t] = a * w2[t];
  __syncthreads();
  for (int s = 32; s > 0; s >>= 1) { if (t < s) red[t] += red[t + s]; __syncthreads(); }
  if (t == 0) out[i] = red[0] + b2[0];
}

// ---------------------------------------------------------------------------
// Hamiltonian assembly
// ---------------------------------------------------------------------------
__global__ void k_dna(const float* __restrict__ x, int* __restrict__ dna, int N) {
  int n = blockIdx.x * blockDim.x + threadIdx.x;
  if (n >= N) return;
  const float* xp = x + n * 4;
  dna[n] = (xp[0] != 0.f || xp[1] != 0.f || xp[2] != 0.f || xp[3] != 0.f) ? 1 : 0;
}

__global__ void k_local(const int* __restrict__ batch, const int* __restrict__ dna,
                        int* __restrict__ localr, int N, int B) {
  int g = threadIdx.x;
  if (g >= B) return;
  int cnt = 0;
  for (int n = 0; n < N; ++n)
    if (batch[n] == g) { localr[n] = cnt; cnt += dna[n]; }
}

__global__ void k_H_init(float* __restrict__ H, int B) {
  int idx = blockIdx.x * blockDim.x + threadIdx.x;
  if (idx >= B * HS * HS) return;
  int r = (idx >> 7) & 127, c = idx & 127;
  H[idx] = (r == c) ? 1e-6f : 0.f;          // fold the 1e-6*I regularizer in
}

__global__ void k_H_diag(float* __restrict__ H, const int* __restrict__ batch,
                         const int* __restrict__ dna, const int* __restrict__ localr,
                         const float* __restrict__ onsite, int N) {
  int n = blockIdx.x * blockDim.x + threadIdx.x;
  if (n >= N || !dna[n]) return;
  int b = batch[n], l = localr[n];
  atomicAdd(&H[b * HS * HS + l * HS + l], onsite[n]);
}

__global__ void k_H_edge(float* __restrict__ H, const int* __restrict__ src,
                         const int* __restrict__ dst, const int* __restrict__ batch,
                         const int* __restrict__ dna, const int* __restrict__ localr,
                         const float* __restrict__ coup, int E) {
  int f = blockIdx.x * blockDim.x + threadIdx.x;
  if (f >= E) return;
  int s = src[f], d = dst[f];
  if (!dna[s] || !dna[d]) return;
  int b = batch[s], u = localr[s], v = localr[d];
  float c = coup[f];
  atomicAdd(&H[b * HS * HS + u * HS + v], c);
  atomicAdd(&H[b * HS * HS + v * HS + u], c);
}

// ---------------------------------------------------------------------------
// NEGF: one block per (energy, graph). H[b] is pulled into LDS via the
// Tensor Data Mover (128x128 f32 2D tile), then two LDS-resident in-place
// Gauss-Jordan inversions, then WMMA matmul Gi = -(Ainv*diag(d)) @ Gr with
// register-resident Gi feeding trace / bilinear reductions.
// ---------------------------------------------------------------------------
__device__ void invert_inplace(float* W, float* col, int tid) {
  for (int k = 0; k < HS; ++k) {
    __syncthreads();
    for (int i = tid; i < HS; i += 256) col[i] = W[i * HS + k];
    __syncthreads();
    float p = 1.0f / col[k];
    for (int j = tid; j < HS; j += 256) W[k * HS + j] *= p;
    __syncthreads();
    for (int idx = tid; idx < HS * HS; idx += 256) {
      int i = idx >> 7, j = idx & 127;
      if (i != k) {
        if (j != k) W[idx] -= col[i] * W[k * HS + j];
        else        W[idx] = -col[i] * p;
      } else if (j == k) {
        W[idx] = p;
      }
    }
    __syncthreads();
  }
}

__global__ __launch_bounds__(256) void negf_kernel(
    const float* __restrict__ H, const float* __restrict__ GL,
    const float* __restrict__ GR, float* __restrict__ T_out,
    float* __restrict__ dos_out, int B) {
  __shared__ float S0[HS * HS];    // A -> Ainv
  __shared__ float S1[HS * HS];    // H -> M -> Gr
  __shared__ float colb[HS];
  __shared__ float dvec[HS];
  __shared__ float red[256];
  int e = blockIdx.x;
  int b = blockIdx.y;
  int tid = threadIdx.x;
  float Eg = -3.0f + 6.0f * (float)e / 99.0f;
  const float* Hb = H + b * HS * HS;

  __builtin_prefetch(&GL[b * HS], 0, 0);
  __builtin_prefetch(&GR[b * HS], 0, 0);

  // ---- bring H[b] into S1 ----
#if HAVE_TDM
  if (tid < 32) {   // single wave issues the TDM descriptor
    unsigned int lds_off = (unsigned int)(size_t)(&S1[0]);
    unsigned long long ga = (unsigned long long)(size_t)Hb;
    v4u g0;
    g0[0] = 1u;                                   // count=1, user descriptor
    g0[1] = lds_off;                              // lds_addr
    g0[2] = (unsigned int)(ga & 0xffffffffu);     // global_addr[31:0]
    g0[3] = (unsigned int)((ga >> 32) & 0x01ffffffu) | 0x80000000u; // [56:32] | type=2
    v8i g1;
    g1[0] = (int)(2u << 16);    // data_size=4B
    g1[1] = (int)(128u << 16);  // tensor_dim0 = 128
    g1[2] = (int)(128u << 16);  // tensor_dim1 = 128
    g1[3] = (int)(128u << 16);  // tile_dim0 = 128
    g1[4] = 128;                // tile_dim1 = 128, tile_dim2 = 0
    g1[5] = 128;                // tensor_dim0_stride = 128
    g1[6] = 0;
    g1[7] = 0;
    v4i gz = {0, 0, 0, 0};
#if __clang_major__ >= 23
    v8i gz8 = {0, 0, 0, 0, 0, 0, 0, 0};
    __builtin_amdgcn_tensor_load_to_lds(g0, g1, gz, gz, gz8, 0);
#else
    __builtin_amdgcn_tensor_load_to_lds(g0, g1, gz, gz, 0);
#endif
    __builtin_amdgcn_s_wait_tensorcnt(0);
  }
#else
  for (int idx = tid; idx < HS * HS; idx += 256) S1[idx] = Hb[idx];
#endif
  for (int i = tid; i < HS; i += 256)
    dvec[i] = 0.5f * (GL[b * HS + i] + GR[b * HS + i]) + 1e-12f;
  __syncthreads();

  // A = Eg*I - H  (H lives in S1)
  for (int idx = tid; idx < HS * HS; idx += 256) {
    int i = idx >> 7, j = idx & 127;
    float v = -S1[idx];
    if (i == j) v += Eg;
    S0[idx] = v;
  }
  __syncthreads();
  invert_inplace(S0, colb, tid);               // S0 = A^-1
  __syncthreads();
  // M = A + Bm Ainv Bm, built in place over H
  for (int idx = tid; idx < HS * HS; idx += 256) {
    int i = idx >> 7, j = idx & 127;
    float a = -S1[idx];
    if (i == j) a += Eg;
    S1[idx] = a + dvec[i] * S0[idx] * dvec[j];
  }
  __syncthreads();
  invert_inplace(S1, colb, tid);               // S1 = Gr
  __syncthreads();

  // Gi tiles via WMMA; accumulate trace(Gi) and gammaL^T (Gr^2+Gi^2) gammaR
  int wave = tid >> 5, lane = tid & 31;
  int half = lane >> 4, l15 = lane & 15;
  float loc_tr = 0.f, loc_T = 0.f;
  for (int t = wave; t < 64; t += 8) {
    int i0 = (t >> 3) << 4;
    int j0 = (t & 7) << 4;
    v8f acc = {};
#pragma unroll
    for (int k0 = 0; k0 < HS; k0 += 32) {
      v16h a, bb;
#pragma unroll
      for (int j = 0; j < 16; ++j) {
        int ka = k0 + klA(j, half);
        a[j]  = (_Float16)(S0[(i0 + l15) * HS + ka] * dvec[ka]);
        bb[j] = (_Float16)S1[(k0 + klB(j, half)) * HS + (j0 + l15)];
      }
      acc = __builtin_amdgcn_wmma_f32_16x16x32_f16(false, a, false, bb,
                                                   (short)0, acc, false, false);
    }
#pragma unroll
    for (int r = 0; r < 8; ++r) {
      int gi_r = i0 + r + half * 8;
      int gi_c = j0 + l15;
      float gi = -acc[r];
      if (gi_r == gi_c) loc_tr += gi;
      float gr = S1[gi_r * HS + gi_c];
      loc_T += GL[b * HS + gi_r] * GR[b * HS + gi_c] * (gr * gr + gi * gi);
    }
  }
  red[tid] = loc_T; __syncthreads();
  for (int s = 128; s > 0; s >>= 1) { if (tid < s) red[tid] += red[tid + s]; __syncthreads(); }
  float Tsum = red[0]; __syncthreads();
  red[tid] = loc_tr; __syncthreads();
  for (int s = 128; s > 0; s >>= 1) { if (tid < s) red[tid] += red[tid + s]; __syncthreads(); }
  if (tid == 0) {
    float tr = red[0];
    T_out[b * NE_PTS + e]   = log10f(fmaxf(Tsum, 1e-16f));
    dos_out[b * NE_PTS + e] = log10f(fmaxf(-tr / 3.14159265358979f, 1e-16f));
  }
}

// ---------------------------------------------------------------------------
extern "C" void kernel_launch(void* const* d_in, const int* in_sizes, int n_in,
                              void* d_out, int out_size, void* d_ws, size_t ws_size,
                              hipStream_t stream) {
  const float* x        = (const float*)d_in[0];
  const int*   eidx     = (const int*)d_in[1];
  const float* eattr    = (const float*)d_in[2];
  const int*   batch    = (const int*)d_in[3];
  const float* GL       = (const float*)d_in[4];
  const float* GR       = (const float*)d_in[5];
  const float* node_w   = (const float*)d_in[6];
  const float* node_b   = (const float*)d_in[7];
  const float* edgep_w  = (const float*)d_in[8];
  const float* edgep_b  = (const float*)d_in[9];
  const float* gat_lin  = (const float*)d_in[10];
  const float* att_src  = (const float*)d_in[11];
  const float* att_dst  = (const float*)d_in[12];
  const float* gat_le   = (const float*)d_in[13];
  const float* att_edge = (const float*)d_in[14];
  const float* gat_bias = (const float*)d_in[15];
  const float* ln_s     = (const float*)d_in[16];
  const float* ln_b     = (const float*)d_in[17];
  const float* on_w1    = (const float*)d_in[18];
  const float* on_b1    = (const float*)d_in[19];
  const float* on_w2    = (const float*)d_in[20];
  const float* on_b2    = (const float*)d_in[21];
  const float* cp_w1    = (const float*)d_in[22];
  const float* cp_b1    = (const float*)d_in[23];
  const float* cp_w2    = (const float*)d_in[24];
  const float* cp_b2    = (const float*)d_in[25];

  const int N = in_sizes[0] / 4;
  const int E = in_sizes[1] / 2;
  const int B = in_sizes[4] / HS;
  const int* src = eidx;
  const int* dst = eidx + E;

  // workspace carve-out
  char* base = (char*)d_ws;
  size_t off = 0;
  auto allocf = [&](size_t n) { float* p = (float*)(base + off); off += n * sizeof(float); return p; };
  float* h      = allocf((size_t)N * HDIM);
  float* ebuf   = allocf((size_t)E * HDIM);
  float* xl     = allocf((size_t)N * HDIM);
  float* el     = allocf((size_t)E * HDIM);
  float* elm    = allocf(HDIM);
  float* ex     = allocf((size_t)(E + N) * NHEAD);
  float* amax   = allocf((size_t)N * NHEAD);
  float* den    = allocf((size_t)N * NHEAD);
  float* hc     = allocf((size_t)N * HDIM);
  float* onsite = allocf(N);
  float* coup   = allocf(E);
  int* dna    = (int*)(base + off); off += (size_t)N * sizeof(int);
  int* localr = (int*)(base + off); off += (size_t)N * sizeof(int);
  (void)ws_size; (void)n_in; (void)out_size;

  float* T_out   = (float*)d_out;
  float* dos_out = T_out + (size_t)B * NE_PTS;
  float* Hout    = T_out + (size_t)2 * B * NE_PTS;

  const int TB = 256;
  k_encode<<<(N * HDIM + TB - 1) / TB, TB, 0, stream>>>(x, node_w, node_b, h, N, 4);
  k_encode<<<(E * HDIM + TB - 1) / TB, TB, 0, stream>>>(eattr, edgep_w, edgep_b, ebuf, E, 5);

  const int F = E + N;
  for (int l = 0; l < NLAYER; ++l) {
    wmma_matmul128<<<dim3((N + 15) / 16), 256, 0, stream>>>(h, gat_lin + (size_t)l * HDIM * HDIM, xl, N);
    wmma_matmul128<<<dim3((E + 15) / 16), 256, 0, stream>>>(ebuf, gat_le + (size_t)l * HDIM * HDIM, el, E);
    k_el_mean<<<1, HDIM, 0, stream>>>(el, elm, E);
    k_zero3<<<(N * HDIM + TB - 1) / TB, TB, 0, stream>>>(amax, den, hc, N);
    k_alpha<<<(F * NHEAD + TB - 1) / TB, TB, 0, stream>>>(
        xl, el, elm, src, dst, att_src + l * HDIM, att_dst + l * HDIM,
        att_edge + l * HDIM, ex, amax, E, N);
    k_expsum<<<(F * NHEAD + TB - 1) / TB, TB, 0, stream>>>(dst, ex, amax, den, E, N);
    k_scatter<<<(F * HDIM + TB - 1) / TB, TB, 0, stream>>>(xl, src, dst, ex, den, hc, E, N);
    k_norm<<<N, HDIM, 0, stream>>>(h, hc, gat_bias + l * HDIM, ln_s + l * HDIM, ln_b + l * HDIM);
  }

  k_mlp<<<N, 64, 0, stream>>>(h, on_w1, on_b1, on_w2, on_b2, onsite);
  k_mlp<<<E, 64, 0, stream>>>(ebuf, cp_w1, cp_b1, cp_w2, cp_b2, coup);
  k_dna<<<(N + TB - 1) / TB, TB, 0, stream>>>(x, dna, N);
  k_local<<<1, 32, 0, stream>>>(batch, dna, localr, N, B);
  k_H_init<<<(B * HS * HS + TB - 1) / TB, TB, 0, stream>>>(Hout, B);
  k_H_diag<<<(N + TB - 1) / TB, TB, 0, stream>>>(Hout, batch, dna, localr, onsite, N);
  k_H_edge<<<(E + TB - 1) / TB, TB, 0, stream>>>(Hout, src, dst, batch, dna, localr, coup, E);

  negf_kernel<<<dim3(NE_PTS, B), 256, 0, stream>>>(Hout, GL, GR, T_out, dos_out, B);
}
